// EnhancedLocalAttention_60885456388529
// MI455X (gfx1250) — matrix operand order repaired and checked by reading, not compiled, not run on hardware
//
#include <hip/hip_runtime.h>

// ---------------------------------------------------------------------------
// EnhancedLocalAttention for MI455X (gfx1250, wave32, WMMA bf16 + TDM)
// B=4, L=4096, C=1024, H=16, D=64, W=512, STRIDE=256
// Only windows n=0..7 contribute to the trimmed output; window n is full
// self-attention over seq slice [n*256, n*256+512). QKV needed only for
// seq rows 0..2303 per batch (LE = 2304 = 7*256+512).
// NOTE: out_gemm is defined first so the disasm snippet shows its TDM loop.
// ---------------------------------------------------------------------------

typedef __attribute__((ext_vector_type(16))) __bf16 v16bf;
typedef __attribute__((ext_vector_type(8)))  float  v8f;
typedef __attribute__((ext_vector_type(4)))  unsigned int v4u;
typedef __attribute__((ext_vector_type(8)))  int v8i;
typedef __attribute__((ext_vector_type(4)))  int v4i;
static_assert(sizeof(v16bf) == 32, "v16bf must be 32 bytes");

union Frag { v16bf v; uint4 q[2]; };

__device__ inline v8f wmma_bf16(const Frag& a, const Frag& b, v8f c) {
  return __builtin_amdgcn_wmma_f32_16x16x32_bf16(
      /*neg_a=*/false, a.v, /*neg_b=*/false, b.v,
      /*c_mod=*/(short)0, c, /*reuse_a=*/false, /*reuse_b=*/false);
}

__device__ inline v8f v8f_zero() {
  v8f z = {0.f, 0.f, 0.f, 0.f, 0.f, 0.f, 0.f, 0.f};
  return z;
}

// round-to-nearest-even f32 -> bf16 bits
__device__ inline unsigned short f2bf(float f) {
  unsigned int u = __float_as_uint(f);
  unsigned int r = u + 0x7fffu + ((u >> 16) & 1u);
  return (unsigned short)(r >> 16);
}
__device__ inline float bf2f(unsigned short h) {
  return __uint_as_float(((unsigned int)h) << 16);
}

constexpr int Bb   = 4;
constexpr int Ls   = 4096;
constexpr int Cc   = 1024;
constexpr int Hh   = 16;
constexpr int Dd   = 64;
constexpr int Ww   = 512;
constexpr int NW   = 8;      // windows that survive the [:L] trim
constexpr int LE   = 2304;   // max seq row used: 7*256+512

#if __has_builtin(__builtin_amdgcn_tensor_load_to_lds)
#define HAVE_TDM 1
#else
#define HAVE_TDM 0
#endif

#if HAVE_TDM
// Issue a 2D TDM tile load: tile_d0 x tile_d1 elements of 2 bytes, row
// stride `stride_el` elements, into LDS at byte offset `lds_off` with
// 16 bytes of LDS padding after every 64 bytes (matches [..][40]-ushort rows).
__device__ inline void tdm_load_tile_2d_bf16(unsigned lds_off, const void* gptr,
                                             unsigned tile_d0, unsigned tile_d1,
                                             unsigned tensor_d0, unsigned tensor_d1,
                                             unsigned stride_el) {
  const unsigned long long ga = (unsigned long long)(size_t)gptr;
  v4u g0;
  g0[0] = 1u;                                  // count=1 valid user descriptor
  g0[1] = lds_off;                             // lds_addr (bytes)
  g0[2] = (unsigned)ga;                        // global_addr[31:0]
  g0[3] = (unsigned)((ga >> 32) & 0x01FFFFFFu) | (2u << 30);  // [56:32] | type=2
  v8i g1;
  // data_size=1 (2B), pad_enable=1, pad_interval=3 (16 DW = 64B),
  // pad_amount=3 (4 DW = 16B); workgroup_mask=0 (not in cluster)
  g1[0] = (int)((1u << 16) | (1u << 20) | (3u << 22) | (3u << 25));
  g1[1] = (int)((tensor_d0 & 0xFFFFu) << 16);                  // td0[15:0] @ bit48
  g1[2] = (int)((tensor_d0 >> 16) | ((tensor_d1 & 0xFFFFu) << 16));
  g1[3] = (int)((tensor_d1 >> 16) | ((tile_d0 & 0xFFFFu) << 16)); // tile_dim0
  g1[4] = (int)(tile_d1 & 0xFFFFu);                            // tile_dim1, dim2=0
  g1[5] = (int)stride_el;                                      // dim0_stride[31:0]
  g1[6] = 0;                                                   // stride hi bits
  g1[7] = 0;
  const v4i gz = {0, 0, 0, 0};
#if defined(__clang_major__) && (__clang_major__ >= 23)
  const v8i gz8 = {0, 0, 0, 0, 0, 0, 0, 0};
  __builtin_amdgcn_tensor_load_to_lds(g0, g1, gz, gz, gz8, 0);
#else
  __builtin_amdgcn_tensor_load_to_lds(g0, g1, gz, gz, 0);
#endif
}
#endif

// ---------------------------------------------------------------------------
// Kernel 3 (defined first): output projection.
// out[m, j] = AO[m, :] @ w_out[:, j] + b_out[j];  M = 16384, N = K = 1024.
// A tile (already bf16, row-major) is staged into LDS by the Tensor Data
// Mover (wave 0), with TDM padding reproducing the 80-byte LDS row stride.
// ---------------------------------------------------------------------------
__global__ __launch_bounds__(256) void out_gemm(
    const unsigned short* __restrict__ AO, const float* __restrict__ w,
    const float* __restrict__ bias, float* __restrict__ out)
{
  constexpr int N = Cc, K = Cc;
  __shared__ unsigned short As[128][40];
  __shared__ unsigned short Bs[128][40];

  const int t = threadIdx.x, lane = t & 31, wave = t >> 5;
  const int mt = blockIdx.y, nt = blockIdx.x;
  const int wm = wave >> 2, wn = wave & 3;

  v8f acc[4][2];
#pragma unroll
  for (int i = 0; i < 4; ++i)
#pragma unroll
    for (int j = 0; j < 2; ++j) acc[i][j] = v8f_zero();

  const int bkr = t >> 5, bnc = (t & 31) * 4;
#if HAVE_TDM
  const unsigned As_lds = (unsigned)(size_t)&As[0][0];
#else
  const int atr = t >> 2, atc8 = (t & 3) * 8;  // fallback A loader
#endif

  for (int k0 = 0; k0 < K; k0 += 32) {
    __syncthreads();
    // ---- stage A tile (128 x 32 bf16) via TDM ----
#if HAVE_TDM
    if (wave == 0) {
      tdm_load_tile_2d_bf16(As_lds, AO + (size_t)(mt * 128) * K + k0,
                            /*tile_d0=*/32, /*tile_d1=*/128,
                            /*tensor_d0=*/(unsigned)K, /*tensor_d1=*/16384u,
                            /*stride_el=*/(unsigned)K);
    }
#else
#pragma unroll
    for (int p = 0; p < 2; ++p) {
      const int row = atr + p * 64;
      *(uint4*)(&As[row][atc8]) =
          *(const uint4*)(AO + (size_t)(mt * 128 + row) * K + k0 + atc8);
    }
#endif
    // ---- stage W tile (32k x 128n f32 -> bf16, transposed to [n][k]) ----
#pragma unroll
    for (int p = 0; p < 4; ++p) {
      const int kk = bkr + p * 8;
      const float4 wv = *(const float4*)(w + (size_t)(k0 + kk) * N + nt * 128 + bnc);
      Bs[bnc + 0][kk] = f2bf(wv.x);
      Bs[bnc + 1][kk] = f2bf(wv.y);
      Bs[bnc + 2][kk] = f2bf(wv.z);
      Bs[bnc + 3][kk] = f2bf(wv.w);
    }
    if (k0 + 32 < K) {
      __builtin_prefetch(w + (size_t)(k0 + 32 + bkr) * N + nt * 128 + bnc, 0, 1);
    }
#if HAVE_TDM
    if (wave == 0) __builtin_amdgcn_s_wait_tensorcnt(0);
#endif
    __syncthreads();

    const int akk = (lane < 16) ? 0 : 8;
    const int bkk = (lane < 16) ? 0 : 8;
    Frag a[4], bf[2];
#pragma unroll
    for (int tt = 0; tt < 4; ++tt) {
      const int row = wm * 64 + tt * 16 + (lane & 15);
      a[tt].q[0] = *(const uint4*)(&As[row][akk]);
      a[tt].q[1] = *(const uint4*)(&As[row][akk + 16]);
    }
#pragma unroll
    for (int ct = 0; ct < 2; ++ct) {
      const int nn = wn * 32 + ct * 16 + (lane & 15);
      bf[ct].q[0] = *(const uint4*)(&Bs[nn][bkk]);
      bf[ct].q[1] = *(const uint4*)(&Bs[nn][bkk + 16]);
    }
#pragma unroll
    for (int tt = 0; tt < 4; ++tt)
#pragma unroll
      for (int ct = 0; ct < 2; ++ct)
        acc[tt][ct] = wmma_bf16(a[tt], bf[ct], acc[tt][ct]);
  }

  const int hi = (lane < 16) ? 0 : 8, col16 = lane & 15;
#pragma unroll
  for (int tt = 0; tt < 4; ++tt)
#pragma unroll
    for (int ct = 0; ct < 2; ++ct)
#pragma unroll
      for (int r = 0; r < 8; ++r) {
        const int m = mt * 128 + wm * 64 + tt * 16 + r + hi;
        const int j = nt * 128 + wn * 32 + ct * 16 + col16;
        out[(size_t)m * N + j] = acc[tt][ct][r] + bias[j];
      }
}

// ---------------------------------------------------------------------------
// Kernel 1: QKV GEMM.  qkv[m, j] = x[m, :] @ w_qkv[:, j] + b_qkv[j]
// M = B*LE = 9216 (trimmed), N = 3072, K = 1024.
// Q scaled by 1/sqrt(D)=0.125 at store. Q,K stored [b][h][seq][d] bf16,
// V stored transposed [b][h][d][seq] bf16 for the PV stage.
// ---------------------------------------------------------------------------
__global__ __launch_bounds__(256) void qkv_gemm(
    const float* __restrict__ x, const float* __restrict__ w,
    const float* __restrict__ bias,
    unsigned short* __restrict__ Qb, unsigned short* __restrict__ Kb,
    unsigned short* __restrict__ Vt)
{
  constexpr int N = 3 * Cc;  // 3072
  constexpr int K = Cc;      // 1024
  __shared__ unsigned short As[128][40];  // [m][k] k-contiguous, pad->40
  __shared__ unsigned short Bs[128][40];  // [n][k] k-contiguous (transposed)

  const int t = threadIdx.x, lane = t & 31, wave = t >> 5;
  const int mt = blockIdx.y, nt = blockIdx.x;
  const int wm = wave >> 2, wn = wave & 3;        // 2x4 wave grid

  v8f acc[4][2];
#pragma unroll
  for (int i = 0; i < 4; ++i)
#pragma unroll
    for (int j = 0; j < 2; ++j) acc[i][j] = v8f_zero();

  const int atr = t >> 3, atc = (t & 7) * 4;      // A loader: 32 rows/pass
  const int bkr = t >> 5, bnc = (t & 31) * 4;     // B loader: 8 k-rows/pass

  for (int k0 = 0; k0 < K; k0 += 32) {
    __syncthreads();
    // ---- stage A tile (128 x 32 f32 -> bf16) ----
#pragma unroll
    for (int p = 0; p < 4; ++p) {
      const int row = atr + p * 32;
      const int m = mt * 128 + row;
      const int batch = m / LE, r = m % LE;
      const float4 xv = *(const float4*)(x + ((size_t)batch * Ls + r) * K + k0 + atc);
      As[row][atc + 0] = f2bf(xv.x);
      As[row][atc + 1] = f2bf(xv.y);
      As[row][atc + 2] = f2bf(xv.z);
      As[row][atc + 3] = f2bf(xv.w);
    }
    // ---- stage W tile (32k x 128n f32 -> bf16, transposed to [n][k]) ----
#pragma unroll
    for (int p = 0; p < 4; ++p) {
      const int kk = bkr + p * 8;
      const float4 wv = *(const float4*)(w + (size_t)(k0 + kk) * N + nt * 128 + bnc);
      Bs[bnc + 0][kk] = f2bf(wv.x);
      Bs[bnc + 1][kk] = f2bf(wv.y);
      Bs[bnc + 2][kk] = f2bf(wv.z);
      Bs[bnc + 3][kk] = f2bf(wv.w);
    }
    if (k0 + 32 < K) {  // prefetch next k-tile (global_prefetch_b8)
      const int m = mt * 128 + atr;
      const int batch = m / LE, r = m % LE;
      __builtin_prefetch(x + ((size_t)batch * Ls + r) * K + k0 + 32 + atc, 0, 1);
      __builtin_prefetch(w + (size_t)(k0 + 32 + bkr) * N + nt * 128 + bnc, 0, 1);
    }
    __syncthreads();

    // ---- fragments + WMMA ----
    const int akk = (lane < 16) ? 0 : 8;   // A: K halves {0-7,16-23}/{8-15,24-31}
    const int bkk = (lane < 16) ? 0 : 8;   // B: K halves {0-7,16-23}/{8-15,24-31}
    Frag a[4], bf[2];
#pragma unroll
    for (int tt = 0; tt < 4; ++tt) {
      const int row = wm * 64 + tt * 16 + (lane & 15);
      a[tt].q[0] = *(const uint4*)(&As[row][akk]);
      a[tt].q[1] = *(const uint4*)(&As[row][akk + 16]);
    }
#pragma unroll
    for (int ct = 0; ct < 2; ++ct) {
      const int nn = wn * 32 + ct * 16 + (lane & 15);
      bf[ct].q[0] = *(const uint4*)(&Bs[nn][bkk]);
      bf[ct].q[1] = *(const uint4*)(&Bs[nn][bkk + 16]);
    }
#pragma unroll
    for (int tt = 0; tt < 4; ++tt)
#pragma unroll
      for (int ct = 0; ct < 2; ++ct)
        acc[tt][ct] = wmma_bf16(a[tt], bf[ct], acc[tt][ct]);
  }

  // ---- scatter-store into Q / K / V^T bf16 buffers ----
  const int hi = (lane < 16) ? 0 : 8, col16 = lane & 15;
#pragma unroll
  for (int tt = 0; tt < 4; ++tt)
#pragma unroll
    for (int ct = 0; ct < 2; ++ct)
#pragma unroll
      for (int r = 0; r < 8; ++r) {
        const int m = mt * 128 + wm * 64 + tt * 16 + r + hi;
        const int j = nt * 128 + wn * 32 + ct * 16 + col16;
        const float v = acc[tt][ct][r] + bias[j];
        const int batch = m / LE, rr = m % LE;
        const int typ = j >> 10, c = j & 1023, h = c >> 6, d = c & 63;
        if (typ == 0) {
          Qb[(((size_t)batch * Hh + h) * LE + rr) * Dd + d] = f2bf(v * 0.125f);
        } else if (typ == 1) {
          Kb[(((size_t)batch * Hh + h) * LE + rr) * Dd + d] = f2bf(v);
        } else {
          Vt[(((size_t)batch * Hh + h) * Dd + d) * LE + rr] = f2bf(v);
        }
      }
}

// ---------------------------------------------------------------------------
// Kernel 2: windowed attention.  64 threads = 2 waves; each wave owns one
// 16-query block of one (b, h, window, 32-query chunk).
// Scores staged in LDS (bf16), softmax stats in f32, PV via WMMA.
// ---------------------------------------------------------------------------
__global__ __launch_bounds__(64) void attn_win(
    const unsigned short* __restrict__ Qb, const unsigned short* __restrict__ Kb,
    const unsigned short* __restrict__ Vt, unsigned short* __restrict__ AO)
{
  __shared__ unsigned short P[2][16][520];  // per-wave 16 x 512 (+8 pad)
  __shared__ float smax[2][16];
  __shared__ float sinv[2][16];

  const int t = threadIdx.x, lane = t & 31, wave = t >> 5;
  int bid = blockIdx.x;
  const int qc = bid & 15; bid >>= 4;   // query chunk (32 queries)
  const int n  = bid & 7;  bid >>= 3;   // window
  const int h  = bid & 15; bid >>= 4;   // head
  const int b  = bid;                   // batch
  const int q0 = qc * 32 + wave * 16;   // query offset inside window
  const int l16 = lane & 15;
  const int hi  = (lane < 16) ? 0 : 8;
  const int akk = (lane < 16) ? 0 : 8;
  const int bkk = (lane < 16) ? 0 : 8;
  const size_t headQK = ((size_t)b * Hh + h) * LE;

  // Q A-fragments (2 chunks over d=64), straight from global (row layout
  // matches the WMMA A fragment exactly).
  const unsigned short* qrow = Qb + (headQK + n * 256 + q0 + l16) * Dd;
  Frag aq[2];
#pragma unroll
  for (int c = 0; c < 2; ++c) {
    aq[c].q[0] = *(const uint4*)(qrow + c * 32 + akk);
    aq[c].q[1] = *(const uint4*)(qrow + c * 32 + akk + 16);
  }

  float mloc[8];
#pragma unroll
  for (int r = 0; r < 8; ++r) mloc[r] = -1e30f;

  // ---- pass 1: S = (Q/sqrt(D)) K^T, staged to LDS, running row-max ----
  for (int kc = 0; kc < 32; ++kc) {
    const unsigned short* krow = Kb + (headQK + n * 256 + kc * 16 + l16) * Dd;
    Frag bk0, bk1;
    bk0.q[0] = *(const uint4*)(krow + bkk);
    bk0.q[1] = *(const uint4*)(krow + bkk + 16);
    bk1.q[0] = *(const uint4*)(krow + 32 + bkk);
    bk1.q[1] = *(const uint4*)(krow + 32 + bkk + 16);
    v8f s = v8f_zero();
    s = wmma_bf16(aq[0], bk0, s);
    s = wmma_bf16(aq[1], bk1, s);
#pragma unroll
    for (int r = 0; r < 8; ++r) {
      const float sv = s[r];
      mloc[r] = fmaxf(mloc[r], sv);
      P[wave][r + hi][kc * 16 + l16] = f2bf(sv);
    }
  }

  // row max across the 16 lanes that share each row
#pragma unroll
  for (int m = 1; m < 16; m <<= 1)
#pragma unroll
    for (int r = 0; r < 8; ++r)
      mloc[r] = fmaxf(mloc[r], __shfl_xor(mloc[r], m, 16));
  if (lane == 0)
#pragma unroll
    for (int r = 0; r < 8; ++r) smax[wave][r] = mloc[r];
  if (lane == 16)
#pragma unroll
    for (int r = 0; r < 8; ++r) smax[wave][8 + r] = mloc[r];
  __syncthreads();

  // ---- exp + row-sum pass (2 lanes per row, 256 cols, 8 values/iter) ----
  {
    const int row = l16;
    const float mx = smax[wave][row];
    unsigned short* prow = &P[wave][row][(lane >> 4) * 256];
    float ssum = 0.f;
    for (int i = 0; i < 32; ++i) {
      uint4 pk = *(uint4*)(prow + i * 8);
      unsigned pv[4] = {pk.x, pk.y, pk.z, pk.w};
#pragma unroll
      for (int j = 0; j < 4; ++j) {
        const float lo = __expf(__uint_as_float(pv[j] << 16) - mx);
        const float hh = __expf(__uint_as_float(pv[j] & 0xFFFF0000u) - mx);
        ssum += lo + hh;
        pv[j] = ((unsigned)f2bf(lo)) | (((unsigned)f2bf(hh)) << 16);
      }
      uint4 st; st.x = pv[0]; st.y = pv[1]; st.z = pv[2]; st.w = pv[3];
      *(uint4*)(prow + i * 8) = st;
    }
    ssum += __shfl_xor(ssum, 16, 32);
    if (lane < 16) sinv[wave][row] = 1.0f / ssum;
  }
  __syncthreads();

  // ---- pass 2: O = P V (P A-frags from LDS, V^T B-frags from global) ----
  v8f acc[4];
#pragma unroll
  for (int dt = 0; dt < 4; ++dt) acc[dt] = v8f_zero();
  const size_t headV = ((size_t)b * Hh + h) * Dd;
  for (int kc = 0; kc < 16; ++kc) {
    Frag ap;
    ap.q[0] = *(const uint4*)(&P[wave][l16][kc * 32 + akk]);
    ap.q[1] = *(const uint4*)(&P[wave][l16][kc * 32 + akk + 16]);
#pragma unroll
    for (int dt = 0; dt < 4; ++dt) {
      const unsigned short* vrow = Vt + (headV + dt * 16 + l16) * LE + n * 256;
      Frag bv;
      bv.q[0] = *(const uint4*)(vrow + kc * 32 + bkk);
      bv.q[1] = *(const uint4*)(vrow + kc * 32 + bkk + 16);
      acc[dt] = wmma_bf16(ap, bv, acc[dt]);
    }
  }

  // ---- normalize + store to attention output (bf16 row-major [B*L][C]) ----
#pragma unroll
  for (int dt = 0; dt < 4; ++dt)
#pragma unroll
    for (int r = 0; r < 8; ++r) {
      const int row = r + hi;
      const float v = acc[dt][r] * sinv[wave][row];
      const size_t orow = (size_t)b * Ls + n * Ww + q0 + row;
      const int col = h * Dd + dt * 16 + l16;
      AO[orow * Cc + col] = f2bf(v);
    }
}

// ---------------------------------------------------------------------------
extern "C" void kernel_launch(void* const* d_in, const int* in_sizes, int n_in,
                              void* d_out, int out_size, void* d_ws, size_t ws_size,
                              hipStream_t stream) {
  (void)in_sizes; (void)n_in; (void)out_size; (void)ws_size;
  const float* x     = (const float*)d_in[0];
  const float* w_qkv = (const float*)d_in[1];
  const float* b_qkv = (const float*)d_in[2];
  const float* w_out = (const float*)d_in[3];
  const float* b_out = (const float*)d_in[4];
  float* out = (float*)d_out;

  char* ws = (char*)d_ws;
  const size_t szQ = (size_t)Bb * Hh * LE * Dd * sizeof(unsigned short);  // 18.9 MB
  unsigned short* Qb = (unsigned short*)(ws);
  unsigned short* Kb = (unsigned short*)(ws + szQ);
  unsigned short* Vt = (unsigned short*)(ws + 2 * szQ);
  unsigned short* AO = (unsigned short*)(ws + 3 * szQ);  // 33.6 MB

  // 1) QKV projection: M=9216 (trimmed), N=3072, K=1024
  qkv_gemm<<<dim3(3 * Cc / 128, Bb * LE / 128), 256, 0, stream>>>(
      x, w_qkv, b_qkv, Qb, Kb, Vt);
  // 2) windowed attention: B*H*NW*(512/32) = 8192 blocks of 2 waves
  attn_win<<<dim3(Bb * Hh * NW * (Ww / 32)), 64, 0, stream>>>(Qb, Kb, Vt, AO);
  // 3) output projection: M=16384, N=K=1024
  out_gemm<<<dim3(Cc / 128, Bb * Ls / 128), 256, 0, stream>>>(AO, w_out, b_out, out);
}
